// STDPLayer_23270132810155
// MI455X (gfx1250) — matrix operand order repaired and checked by reading, not compiled, not run on hardware
//
#include <hip/hip_runtime.h>
#include <math.h>

// ---------------------------------------------------------------------------
// STDP layer for MI455X (gfx1250): two batch-reduction GEMMs (dw) + one
// forward GEMM (current), all on V_WMMA_F32_16X16X32_F16, wave32.
// Each wave computes a 32x64 output tile: A operands loaded once per K-step
// and reused over 4 B-tiles -> 1.5 b128 loads per WMMA (operands L2-resident).
// ---------------------------------------------------------------------------

typedef _Float16 half_t;
typedef __attribute__((ext_vector_type(16))) _Float16 v16h;
typedef __attribute__((ext_vector_type(8)))  _Float16 v8h;
typedef __attribute__((ext_vector_type(8)))  float    v8f;

constexpr int BATCH  = 1024;
constexpr int PRE_N  = 4096;
constexpr int POST_N = 4096;

constexpr float TAU        = 20.0f;
constexpr float A_PLUS     = 0.01f;
constexpr float A_MINUS    = 0.01f;
constexpr float W_MAXC     = 1.0f;
constexpr float W_MINC     = 0.0f;
constexpr float TARGET     = 0.1f;
constexpr float HOMEO_C    = 0.001f;
constexpr float RATE_ALPHA = 0.01f;
constexpr float DT_C       = 1.0f;

// ---------------------------------------------------------------------------
// WMMA helpers (CDNA5 ISA 7.12.2, 16-bit operands, wave32):
//  A (16xK, K-block 32): lane L holds row M=L%16; halves = two 8-half runs at
//    k0 + 8*(L>=16) and k0 + 16 + 8*(L>=16).
//  B (Kx16): lane L holds column N=L%16; same two 8-half runs as A.
//  C/D f32: VGPR r, lane L -> row M = r + 8*(L>=16), col N = L%16.
// ---------------------------------------------------------------------------

__device__ __forceinline__ v8f wmma_f16(v16h a, v16h b, v8f c) {
  return __builtin_amdgcn_wmma_f32_16x16x32_f16(
      /*neg_a=*/false, a, /*neg_b=*/false, b,
      /*c_mod=*/(short)0, c, /*reuse_a=*/false, /*reuse_b=*/false);
}

__device__ __forceinline__ v16h load_a16(const half_t* __restrict__ row, int k0, int hi) {
  const half_t* p = row + k0 + (hi << 3);
  v8h lo = *(const v8h*)(p);        // k0+8hi .. +7
  v8h hh = *(const v8h*)(p + 16);   // k0+16+8hi .. +7
  return __builtin_shufflevector(lo, hh, 0,1,2,3,4,5,6,7,8,9,10,11,12,13,14,15);
}

__device__ __forceinline__ v16h load_b16(const half_t* __restrict__ row, int k0, int hi) {
  const half_t* p = row + k0 + (hi << 3);
  v8h lo = *(const v8h*)(p);
  v8h hh = *(const v8h*)(p + 16);
  return __builtin_shufflevector(lo, hh, 0,1,2,3,4,5,6,7,8,9,10,11,12,13,14,15);
}

// ---------------------------------------------------------------------------
// Elementwise: new_trace = trace*decay + spikes ; optional f16 copy of spikes.
// ---------------------------------------------------------------------------
__global__ void trace_kernel(const float* __restrict__ spikes,
                             const float* __restrict__ trace,
                             float* __restrict__ new_trace,
                             half_t* __restrict__ spikes_h,
                             float decay) {
  size_t i = (size_t)blockIdx.x * blockDim.x + threadIdx.x;
  float s = spikes[i];
  new_trace[i] = trace[i] * decay + s;
  if (spikes_h) spikes_h[i] = (half_t)s;
}

// ---------------------------------------------------------------------------
// Tiled transpose with scale: src f32 (rows x cols) -> dst f16 (cols x rows).
// sqrt-split STDP scales folded in so both dw terms share one accumulator.
// ---------------------------------------------------------------------------
__global__ void transpose_scale_kernel(const float* __restrict__ src,
                                       half_t* __restrict__ dst,
                                       int rows, int cols, float scale) {
  __shared__ half_t tile[32][33];
  int c0 = blockIdx.x * 32;
  int r0 = blockIdx.y * 32;
  int tx = threadIdx.x, ty = threadIdx.y;
#pragma unroll
  for (int j = 0; j < 4; ++j) {
    int r = r0 + ty + 8 * j;
    tile[ty + 8 * j][tx] = (half_t)(src[(size_t)r * cols + c0 + tx] * scale);
  }
  __syncthreads();
#pragma unroll
  for (int j = 0; j < 4; ++j) {
    int c = c0 + ty + 8 * j;
    dst[(size_t)c * rows + r0 + tx] = tile[tx][ty + 8 * j];
  }
}

// ---------------------------------------------------------------------------
// Rate EMA: new_rate = rate*(1-a) + a*mean_b(post_spikes).
// ---------------------------------------------------------------------------
__global__ void rate_kernel(const float* __restrict__ post_spikes,
                            const float* __restrict__ rate,
                            float* __restrict__ new_rate) {
  int p = blockIdx.x * blockDim.x + threadIdx.x;
  float s = 0.f;
  for (int b = 0; b < BATCH; ++b) s += post_spikes[(size_t)b * POST_N + p];
  new_rate[p] = rate[p] * (1.f - RATE_ALPHA) + RATE_ALPHA * (s * (1.f / BATCH));
}

// ---------------------------------------------------------------------------
// dw GEMM: W_new[p,q] = clip(W + sum_b aP[p,b]*bP[q,b] + aM[p,b]*bM[q,b]
//                              + homeo[p], 0, 1)
// aP = postS^T * sP, bP = preTrace^T * (decay*sP)   (plus term)
// aM = postT^T * (-sM*decay), bM = preS^T * sM      (minus term)
// One wave: 32x64 tile; per K-step: 8 A-loads + 16 B-loads feed 16 WMMAs.
// ---------------------------------------------------------------------------
__global__ void __launch_bounds__(128)
stdp_dw_wmma(const half_t* __restrict__ postS_T,  // (POST, B)
             const half_t* __restrict__ postT_T,  // (POST, B)
             const half_t* __restrict__ preT_T,   // (PRE,  B)
             const half_t* __restrict__ preS_T,   // (PRE,  B)
             const float*  __restrict__ W,
             const float*  __restrict__ rate,
             float*  __restrict__ Wout,
             half_t* __restrict__ Wh) {
  const int lane  = threadIdx.x & 31;
  const int wave  = blockIdx.x * (blockDim.x >> 5) + (threadIdx.x >> 5);
  const int QW    = PRE_N / 64;               // 64 q-strips
  const int p0    = (wave / QW) * 32;         // 32 output rows per wave
  const int q0    = (wave % QW) * 64;         // 64 output cols per wave
  const int lm    = lane & 15;
  const int hi    = lane >> 4;

  const half_t* aP0 = postS_T + (size_t)(p0 + lm) * BATCH;
  const half_t* aM0 = postT_T + (size_t)(p0 + lm) * BATCH;
  const half_t* aP1 = aP0 + (size_t)16 * BATCH;
  const half_t* aM1 = aM0 + (size_t)16 * BATCH;
  const half_t* bP  = preT_T  + (size_t)(q0 + lm) * BATCH;
  const half_t* bM  = preS_T  + (size_t)(q0 + lm) * BATCH;

  v8f acc[2][4] = {};
  for (int kb = 0; kb < BATCH; kb += 32) {
    v16h AP0 = load_a16(aP0, kb, hi);
    v16h AM0 = load_a16(aM0, kb, hi);
    v16h AP1 = load_a16(aP1, kb, hi);
    v16h AM1 = load_a16(aM1, kb, hi);
#pragma unroll
    for (int n = 0; n < 4; ++n) {
      v16h BP = load_b16(bP + (size_t)(16 * n) * BATCH, kb, hi);
      v16h BM = load_b16(bM + (size_t)(16 * n) * BATCH, kb, hi);
      acc[0][n] = wmma_f16(AP0, BP, acc[0][n]);
      acc[0][n] = wmma_f16(AM0, BM, acc[0][n]);
      acc[1][n] = wmma_f16(AP1, BP, acc[1][n]);
      acc[1][n] = wmma_f16(AM1, BM, acc[1][n]);
    }
  }

#pragma unroll
  for (int t = 0; t < 2; ++t) {
    float homeo[8];
#pragma unroll
    for (int r = 0; r < 8; ++r)
      homeo[r] = -HOMEO_C * (rate[p0 + 16 * t + r + 8 * hi] - TARGET);
#pragma unroll
    for (int n = 0; n < 4; ++n) {
      int q = q0 + 16 * n + lm;
#pragma unroll
      for (int r = 0; r < 8; ++r) {
        size_t idx = (size_t)(p0 + 16 * t + r + 8 * hi) * PRE_N + q;
        float w = W[idx] + acc[t][n][r] + homeo[r];
        w = fminf(fmaxf(w, W_MINC), W_MAXC);
        Wout[idx] = w;
        Wh[idx]   = (half_t)w;
      }
    }
  }
}

// ---------------------------------------------------------------------------
// current GEMM: current[b,p] = sum_q preS[b,q] * W_new[p,q]   (K = 4096)
// A = preS_h row-major (B x PRE); B-operand lane holds column p -> row p of
// W_h (row-major), contiguous in q. 32x64 tile per wave, 128 K-steps.
// ---------------------------------------------------------------------------
__global__ void __launch_bounds__(128)
stdp_current_wmma(const half_t* __restrict__ preS_h,  // (B, PRE)
                  const half_t* __restrict__ Wh,      // (POST, PRE)
                  float* __restrict__ current) {      // (B, POST)
  const int lane = threadIdx.x & 31;
  const int wave = blockIdx.x * (blockDim.x >> 5) + (threadIdx.x >> 5);
  const int PW   = POST_N / 64;               // 64 p-strips
  const int b0   = (wave / PW) * 32;
  const int p0   = (wave % PW) * 64;
  const int lm   = lane & 15;
  const int hi   = lane >> 4;

  const half_t* a0 = preS_h + (size_t)(b0 + lm) * PRE_N;
  const half_t* a1 = a0 + (size_t)16 * PRE_N;
  const half_t* br = Wh + (size_t)(p0 + lm) * PRE_N;

  v8f acc[2][4] = {};
  for (int kb = 0; kb < PRE_N; kb += 32) {
    v16h A0 = load_a16(a0, kb, hi);
    v16h A1 = load_a16(a1, kb, hi);
#pragma unroll
    for (int n = 0; n < 4; ++n) {
      v16h Bn = load_b16(br + (size_t)(16 * n) * PRE_N, kb, hi);
      acc[0][n] = wmma_f16(A0, Bn, acc[0][n]);
      acc[1][n] = wmma_f16(A1, Bn, acc[1][n]);
    }
  }

#pragma unroll
  for (int t = 0; t < 2; ++t) {
#pragma unroll
    for (int n = 0; n < 4; ++n) {
      int p = p0 + 16 * n + lm;
#pragma unroll
      for (int r = 0; r < 8; ++r) {
        int b = b0 + 16 * t + r + 8 * hi;
        current[(size_t)b * POST_N + p] = acc[t][n][r];
      }
    }
  }
}

// ---------------------------------------------------------------------------
extern "C" void kernel_launch(void* const* d_in, const int* in_sizes, int n_in,
                              void* d_out, int out_size, void* d_ws, size_t ws_size,
                              hipStream_t stream) {
  (void)in_sizes; (void)n_in; (void)out_size;

  const float* pre_spikes   = (const float*)d_in[0];  // (B, PRE)
  const float* post_spikes  = (const float*)d_in[1];  // (B, POST)
  const float* weights      = (const float*)d_in[2];  // (POST, PRE)
  const float* pre_trace    = (const float*)d_in[3];  // (B, PRE)
  const float* post_trace   = (const float*)d_in[4];  // (B, POST)
  const float* running_rate = (const float*)d_in[5];  // (POST,)

  constexpr size_t nBP = (size_t)BATCH * PRE_N;       // == B*POST too
  constexpr size_t nW  = (size_t)POST_N * PRE_N;

  float* current      = (float*)d_out;                // out 0
  float* new_weights  = current + nBP;                // out 1
  float* new_pre_tr   = new_weights + nW;             // out 2
  float* new_post_tr  = new_pre_tr + nBP;             // out 3
  float* new_rate     = new_post_tr + nBP;            // out 4

  // Workspace layout: 5 * 8 MiB f16 operand buffers + 32 MiB f16 weights.
  constexpr size_t HBP = nBP * sizeof(half_t);        // 8 MiB
  if (ws_size < 5 * HBP + nW * sizeof(half_t)) return;
  char*  ws      = (char*)d_ws;
  half_t* preS_h  = (half_t*)(ws);                    // (B, PRE) unscaled
  half_t* preS_T  = (half_t*)(ws + 1 * HBP);          // (PRE, B) * sM
  half_t* preT_T  = (half_t*)(ws + 2 * HBP);          // (PRE, B) * sP*decay
  half_t* postS_T = (half_t*)(ws + 3 * HBP);          // (POST, B) * sP
  half_t* postT_T = (half_t*)(ws + 4 * HBP);          // (POST, B) * -sM*decay
  half_t* Wh      = (half_t*)(ws + 5 * HBP);          // (POST, PRE)

  const float decay = expf(-DT_C / TAU);              // same for both taus
  const float sP = sqrtf(A_PLUS  / (float)BATCH);     // 0.003125 -> sP*sP = A+/B
  const float sM = sqrtf(A_MINUS / (float)BATCH);

  // 1) traces (+ f16 copy of pre_spikes for the current GEMM)
  trace_kernel<<<(unsigned)(nBP / 256), 256, 0, stream>>>(
      pre_spikes, pre_trace, new_pre_tr, preS_h, decay);
  trace_kernel<<<(unsigned)(nBP / 256), 256, 0, stream>>>(
      post_spikes, post_trace, new_post_tr, nullptr, decay);

  // 2) scaled f16 transposes (K-major operands for the batch-reduction GEMMs)
  dim3 tb(32, 8), tg(PRE_N / 32, BATCH / 32);
  transpose_scale_kernel<<<tg, tb, 0, stream>>>(pre_spikes,  preS_T,  BATCH, PRE_N,  sM);
  transpose_scale_kernel<<<tg, tb, 0, stream>>>(pre_trace,   preT_T,  BATCH, PRE_N,  sP * decay);
  transpose_scale_kernel<<<tg, tb, 0, stream>>>(post_spikes, postS_T, BATCH, POST_N, sP);
  transpose_scale_kernel<<<tg, tb, 0, stream>>>(post_trace,  postT_T, BATCH, POST_N, -sM * decay);

  // 3) homeostatic rate EMA (independent; uses OLD running_rate)
  rate_kernel<<<POST_N / 256, 256, 0, stream>>>(post_spikes, running_rate, new_rate);

  // 4) fused STDP weight update GEMM (32x64 tile/wave)
  stdp_dw_wmma<<<(POST_N / 32) * (PRE_N / 64) / 4, 128, 0, stream>>>(
      postS_T, postT_T, preT_T, preS_T, weights, running_rate, new_weights, Wh);

  // 5) forward current GEMM with updated weights (32x64 tile/wave)
  stdp_current_wmma<<<(BATCH / 32) * (POST_N / 64) / 4, 128, 0, stream>>>(
      preS_h, Wh, current);
}
